// TrajectoryAttention_58789512348461
// MI455X (gfx1250) — compile-verified, matrix operand order checked by reading, not run on hardware
//
#include <hip/hip_runtime.h>
#include <stdint.h>

typedef unsigned short u16;
typedef __attribute__((ext_vector_type(16))) __bf16 v16bf;
typedef __attribute__((ext_vector_type(8)))  float  v8f;

#define BB 4
#define PP 196
#define FF 8
#define CC 768
#define NN 1569   /* 1 + F*P */
#define SS 1568   /* F*P */
#define HH 12
#define DD 64
#define BH (BB*HH)
#define PPAD 224  /* P padded to multiple of 32 for K-dim of P@V */

struct U8x32 { uint4 lo, hi; };

__device__ __forceinline__ u16 f2bf(float f) {
  unsigned u = __float_as_uint(f);
  u += 0x7FFFu + ((u >> 16) & 1u);
  return (u16)(u >> 16);
}
__device__ __forceinline__ float bf2f(u16 h) {
  return __uint_as_float(((unsigned)h) << 16);
}
__device__ __forceinline__ v8f vzero() {
  v8f z = {0.f,0.f,0.f,0.f,0.f,0.f,0.f,0.f};
  return z;
}
// Load one 16x32 bf16 WMMA fragment half per lane: two contiguous 16B chunks.
// lanes 0-15 take K {k..k+7, k+16..k+23}; lanes 16-31 take {k+8..k+15, k+24..k+31}
// (caller passes pointers already offset by kh = (lane>>4)*8).
__device__ __forceinline__ v16bf ld_frag(const u16* lo, const u16* hi) {
  U8x32 t;
  t.lo = *(const uint4*)lo;
  t.hi = *(const uint4*)hi;
  return __builtin_bit_cast(v16bf, t);
}

// CDNA5 async cache->LDS copy (no VGPR round trip), tracked by ASYNCcnt.
// ldsOff: per-lane LDS byte address; gsrc: per-lane 64-bit global address.
__device__ __forceinline__ void async_ld128(unsigned ldsOff, const void* gsrc) {
  asm volatile("global_load_async_to_lds_b128 %0, %1, off"
               :: "v"(ldsOff), "v"(gsrc) : "memory");
}
__device__ __forceinline__ void wait_asynccnt0() {
  asm volatile("s_wait_asynccnt 0" ::: "memory");
}

// ---- generic wave GEMM: one wave -> 16(M) x 64(N) f32 tile, A and B^T row-major bf16, K mult of 32
__device__ __forceinline__ void wmma_gemm4(const u16* __restrict__ A,
                                           const u16* __restrict__ BT,
                                           int K, long aRow0, long aRowMax,
                                           long bRow0, v8f acc[4]) {
  const int lane  = threadIdx.x & 31;
  const int laneM = lane & 15;
  const int kh    = (lane >> 4) << 3;
  long ar = aRow0 + laneM; if (ar > aRowMax) ar = aRowMax;
  const u16* Ap = A + (size_t)ar * K;
  const u16* Bp[4];
#pragma unroll
  for (int t = 0; t < 4; ++t)
    Bp[t] = BT + (size_t)(bRow0 + t*16 + laneM) * K;
  // pull the streams toward the WGP (lowers to global_prefetch_b8)
  __builtin_prefetch(Ap, 0, 1);
#pragma unroll
  for (int t = 0; t < 4; ++t) __builtin_prefetch(Bp[t], 0, 1);
  for (int k = 0; k < K; k += 32) {
    v16bf a = ld_frag(Ap + k + kh, Ap + k + 16 + kh);
#pragma unroll
    for (int t = 0; t < 4; ++t) {
      v16bf b = ld_frag(Bp[t] + k + kh, Bp[t] + k + 16 + kh);
      acc[t] = __builtin_amdgcn_wmma_f32_16x16x32_bf16(false, a, false, b,
                                                       (short)0, acc[t], false, false);
    }
  }
}

// ---------------- conversion / transpose helpers ----------------
__global__ void k_f2bf(const float* __restrict__ in, u16* __restrict__ out, long n) {
  for (long i = (long)blockIdx.x*blockDim.x + threadIdx.x; i < n;
       i += (long)gridDim.x*blockDim.x)
    out[i] = f2bf(in[i]);
}
// out[j*K + c] = in[c*ldIn + j]   (weight -> B^T row-major-K bf16)
__global__ void k_transpose_bf(const float* __restrict__ in, u16* __restrict__ out,
                               int K, int N0, int ldIn) {
  long n = (long)K * N0;
  for (long i = (long)blockIdx.x*blockDim.x + threadIdx.x; i < n;
       i += (long)gridDim.x*blockDim.x) {
    long j = i / K, c = i % K;
    out[(size_t)j*K + c] = f2bf(in[(size_t)c*ldIn + j]);
  }
}

// ---------------- QKV projection, scatter to head-major bf16 ----------------
__global__ __launch_bounds__(128) void k_gemm_qkv(const u16* __restrict__ xb,
    const u16* __restrict__ wqkvT, u16* __restrict__ qh, u16* __restrict__ khv,
    u16* __restrict__ vh) {
  const int wave = threadIdx.x >> 5;
  const int lane = threadIdx.x & 31;
  const int laneM = lane & 15, g = lane >> 4;
  long m0 = (long)blockIdx.x * 64 + wave * 16;
  long n0 = (long)blockIdx.y * 64;
  v8f acc[4] = {vzero(), vzero(), vzero(), vzero()};
  wmma_gemm4(xb, wqkvT, CC, m0, (long)BB*NN - 1, n0, acc);
#pragma unroll
  for (int t = 0; t < 4; ++t) {
    int col = (int)n0 + t*16 + laneM;
    int which = col / CC, jj = col % CC;
    int head = jj / DD, dd = jj % DD;
    u16* base = (which == 0) ? qh : ((which == 1) ? khv : vh);
#pragma unroll
    for (int r = 0; r < 8; ++r) {
      long m = m0 + r + 8*g;
      if (m < (long)BB*NN) {
        int b = (int)(m / NN), n = (int)(m % NN);
        base[((size_t)(b*HH + head)*NN + n)*DD + dd] = f2bf(acc[t][r]);
      }
    }
  }
}

// ---------------- cls-token attention (1 query per head) ----------------
__global__ __launch_bounds__(256) void k_cls(const u16* __restrict__ qh,
    const u16* __restrict__ khv, const u16* __restrict__ vh, u16* __restrict__ pre) {
  __shared__ float qs[DD];
  __shared__ float sc[NN];
  __shared__ float red[256];
  const int tid = threadIdx.x;
  const int bh = blockIdx.x;
  const int b = bh / HH, head = bh % HH;
  const size_t base = (size_t)bh * NN * DD;
  if (tid < DD) qs[tid] = bf2f(qh[base + tid]);
  __syncthreads();
  float lmax = -1e30f;
  for (int j = tid; j < NN; j += 256) {
    float a = 0.f;
    const u16* kp = khv + base + (size_t)j * DD;
#pragma unroll 8
    for (int dd = 0; dd < DD; ++dd) a += qs[dd] * bf2f(kp[dd]);
    a *= 0.125f;
    sc[j] = a; lmax = fmaxf(lmax, a);
  }
  red[tid] = lmax; __syncthreads();
  for (int s = 128; s > 0; s >>= 1) { if (tid < s) red[tid] = fmaxf(red[tid], red[tid+s]); __syncthreads(); }
  float mx = red[0]; __syncthreads();
  float lsum = 0.f;
  for (int j = tid; j < NN; j += 256) { float e = __expf(sc[j] - mx); sc[j] = e; lsum += e; }
  red[tid] = lsum; __syncthreads();
  for (int s = 128; s > 0; s >>= 1) { if (tid < s) red[tid] += red[tid+s]; __syncthreads(); }
  float is = 1.f / red[0];
  if (tid < DD) {
    float acc = 0.f;
    for (int j = 0; j < NN; ++j) acc += sc[j] * bf2f(vh[base + (size_t)j * DD + tid]);
    pre[((size_t)b * NN) * CC + head * DD + tid] = f2bf(acc * is);
  }
}

// ---------------- flash-style space attention per (head, frame) ----------------
// LDS plan (58 KB total):
//   regionA (28,672 B): phase 1 = raw V tile (async-copied, row-major key x d)
//                       phase 2 = per-wave probabilities pr[4][16][PPAD]
//   vt      (29,696 B): V^T (d x key), zero-padded keys
__global__ __launch_bounds__(128) void k_space_attn(const u16* __restrict__ qh,
    const u16* __restrict__ khv, const u16* __restrict__ vh, u16* __restrict__ xt) {
  __shared__ __align__(16) u16 regionA[4*16*PPAD];   // 28,672 B dual-use
  __shared__ __align__(16) u16 vt[DD][PPAD + 8];     // V^T staged once per block
  const int tid = threadIdx.x;
  const int wave = tid >> 5;
  const int lane = tid & 31;
  const int laneM = lane & 15, g = lane >> 4;
  const int kh = g << 3;
  const int f = blockIdx.y;
  const int bh = blockIdx.z;
  const int b = bh / HH, head = bh % HH;
  const int q0 = blockIdx.x * 64 + wave * 16;

  // phase 1a: async DMA the valid V tile (196 keys x 64 d, bf16) into regionA
  const size_t vbase = ((size_t)bh * NN + 1 + (size_t)f * PP) * DD;
  {
    const char* gsrc = (const char*)(vh + vbase);
    unsigned lds0 = (unsigned)(uintptr_t)(&regionA[0]);
    const int n128 = (PP * DD * 2) / 16;  // 1568 x 16B transfers
    for (int idx = tid; idx < n128; idx += 128)
      async_ld128(lds0 + idx*16, gsrc + idx*16);
    wait_asynccnt0();
  }
  __syncthreads();
  // phase 1b: LDS->LDS transpose into vt, zero-fill padded keys
  {
    const u16 (*vraw)[DD] = (const u16(*)[DD])regionA;
    for (int idx = tid; idx < PPAD * DD; idx += 128) {
      int p = idx / DD, dd = idx % DD;
      vt[dd][p] = (p < PP) ? vraw[p][dd] : (u16)0;
    }
  }
  __syncthreads();

  // scores: Q(16x64) x K^T -> 16x224
  long qrow = (long)bh * NN + 1 + q0 + laneM;
  long qmax = (long)bh * NN + SS;
  if (qrow > qmax) qrow = qmax;
  const u16* Qp = qh + (size_t)qrow * DD;
  const u16* Kp[14];
#pragma unroll
  for (int t = 0; t < 14; ++t) {
    int key = t*16 + laneM; if (key > PP-1) key = PP-1;
    Kp[t] = khv + ((size_t)bh * NN + 1 + (size_t)f * PP + key) * DD;
  }
  v8f sc[14];
#pragma unroll
  for (int t = 0; t < 14; ++t) sc[t] = vzero();
#pragma unroll
  for (int k = 0; k < DD; k += 32) {
    v16bf a = ld_frag(Qp + k + kh, Qp + k + 16 + kh);
#pragma unroll
    for (int t = 0; t < 14; ++t) {
      v16bf bfr = ld_frag(Kp[t] + k + kh, Kp[t] + k + 16 + kh);
      sc[t] = __builtin_amdgcn_wmma_f32_16x16x32_bf16(false, a, false, bfr,
                                                      (short)0, sc[t], false, false);
    }
  }
#pragma unroll
  for (int t = 0; t < 14; ++t) sc[t] *= 0.125f;
#pragma unroll
  for (int t = 12; t < 14; ++t) {
    if (t*16 + laneM >= PP) {
#pragma unroll
      for (int r = 0; r < 8; ++r) sc[t][r] = -1e30f;
    }
  }
  // per-row softmax: row m=r+8g lives on the 16 lanes of half g; xor-reduce within half
  u16 (*pr)[16][PPAD] = (u16(*)[16][PPAD])regionA;
  for (int r = 0; r < 8; ++r) {
    float mx = -1e30f;
#pragma unroll
    for (int t = 0; t < 14; ++t) mx = fmaxf(mx, sc[t][r]);
#pragma unroll
    for (int m = 1; m < 16; m <<= 1) mx = fmaxf(mx, __shfl_xor(mx, m, 32));
    float e[14], sum = 0.f;
#pragma unroll
    for (int t = 0; t < 14; ++t) { e[t] = __expf(sc[t][r] - mx); sum += e[t]; }
#pragma unroll
    for (int m = 1; m < 16; m <<= 1) sum += __shfl_xor(sum, m, 32);
    float is = 1.f / sum;
    int row = r + 8*g;
#pragma unroll
    for (int t = 0; t < 14; ++t) pr[wave][row][t*16 + laneM] = f2bf(e[t] * is);
  }
  __syncthreads();

  // out: P(16x224) x V(224x64)
  v8f o[4] = {vzero(), vzero(), vzero(), vzero()};
  const u16* Pr = &pr[wave][laneM][0];
  const u16* Vr[4];
#pragma unroll
  for (int t = 0; t < 4; ++t) Vr[t] = &vt[t*16 + laneM][0];
  for (int k = 0; k < PPAD; k += 32) {
    v16bf a = ld_frag(Pr + k + kh, Pr + k + 16 + kh);
#pragma unroll
    for (int t = 0; t < 4; ++t) {
      v16bf bfr = ld_frag(Vr[t] + k + kh, Vr[t] + k + 16 + kh);
      o[t] = __builtin_amdgcn_wmma_f32_16x16x32_bf16(false, a, false, bfr,
                                                     (short)0, o[t], false, false);
    }
  }
  // xt[b, q, f, head*64+dd] bf16
#pragma unroll
  for (int r = 0; r < 8; ++r) {
    int q = q0 + r + 8*g;
    if (q < SS) {
      size_t base = (((size_t)(b * SS + q) * FF) + f) * CC + head * DD;
#pragma unroll
      for (int t = 0; t < 4; ++t)
        xt[base + t*16 + laneM] = f2bf(o[t][r]);
    }
  }
}

// ---------------- diagonal extract: x_diag[b,s,:] = xt[b,s,s/P,:] ----------------
__global__ void k_diag(const u16* __restrict__ xt, u16* __restrict__ dg) {
  const unsigned total = BB * SS * (CC/2);  // dword copies
  for (unsigned i = blockIdx.x*blockDim.x + threadIdx.x; i < total;
       i += gridDim.x*blockDim.x) {
    unsigned row = i / (CC/2), cp = i % (CC/2);
    unsigned s = row % SS;
    unsigned fsel = s / PP;
    ((unsigned*)dg)[(size_t)row*(CC/2) + cp] =
      ((const unsigned*)xt)[(((size_t)row*FF) + fsel)*(CC/2) + cp];
  }
}

// ---------------- q2 = (x_diag @ w_q) * scale ----------------
__global__ __launch_bounds__(128) void k_gemm_q2(const u16* __restrict__ dg,
    const u16* __restrict__ wqT, u16* __restrict__ q2) {
  const int wave = threadIdx.x >> 5;
  const int lane = threadIdx.x & 31;
  const int laneM = lane & 15, g = lane >> 4;
  long m0 = (long)blockIdx.x*64 + wave*16;
  long n0 = (long)blockIdx.y*64;
  v8f acc[4] = {vzero(),vzero(),vzero(),vzero()};
  wmma_gemm4(dg, wqT, CC, m0, (long)BB*SS - 1, n0, acc);
#pragma unroll
  for (int t = 0; t < 4; ++t) {
    int col = (int)n0 + t*16 + laneM;
#pragma unroll
    for (int r = 0; r < 8; ++r) {
      long m = m0 + r + 8*g;
      if (m < (long)BB*SS) q2[(size_t)m*CC + col] = f2bf(acc[t][r] * 0.125f);
    }
  }
}

// ---------------- k2 = xt @ w_kv[:, :C] ----------------
__global__ __launch_bounds__(128) void k_gemm_k2(const u16* __restrict__ A,
    const u16* __restrict__ wkT, u16* __restrict__ k2) {
  const int wave = threadIdx.x >> 5;
  const int lane = threadIdx.x & 31;
  const int laneM = lane & 15, g = lane >> 4;
  long m0 = (long)blockIdx.x*64 + wave*16;
  long n0 = (long)blockIdx.y*64;
  v8f acc[4] = {vzero(),vzero(),vzero(),vzero()};
  wmma_gemm4(A, wkT, CC, m0, (long)BB*SS*FF - 1, n0, acc);
#pragma unroll
  for (int t = 0; t < 4; ++t) {
    int col = (int)n0 + t*16 + laneM;
#pragma unroll
    for (int r = 0; r < 8; ++r) {
      long m = m0 + r + 8*g;
      k2[(size_t)m*CC + col] = f2bf(acc[t][r]);
    }
  }
}

// ---------------- attn2 softmax over F + frame-weighted output ----------------
__global__ __launch_bounds__(256) void k_attn2(const u16* __restrict__ q2,
    const u16* __restrict__ k2, const u16* __restrict__ xt,
    u16* __restrict__ pre, float* __restrict__ attn2_out) {
  int idx = blockIdx.x*blockDim.x + threadIdx.x;
  const int total = BB*HH*SS;
  if (idx >= total) return;
  int s = idx % SS; int h = (idx / SS) % HH; int b = idx / (SS*HH);
  const u16* qp = q2 + ((size_t)(b*SS + s))*CC + h*DD;
  float qv[DD];
#pragma unroll 8
  for (int dd = 0; dd < DD; ++dd) qv[dd] = bf2f(qp[dd]);
  float e[FF]; float mx = -1e30f;
  for (int f = 0; f < FF; ++f) {
    const u16* kp = k2 + (((size_t)(b*SS + s))*FF + f)*CC + h*DD;
    float a = 0.f;
#pragma unroll 8
    for (int dd = 0; dd < DD; ++dd) a += qv[dd]*bf2f(kp[dd]);
    e[f] = a; mx = fmaxf(mx, a);
  }
  float sum = 0.f;
#pragma unroll
  for (int f = 0; f < FF; ++f) { e[f] = __expf(e[f] - mx); sum += e[f]; }
  float is = 1.f / sum;
  float* ao = attn2_out + (((size_t)(b*HH + h))*SS + s)*FF;
#pragma unroll
  for (int f = 0; f < FF; ++f) ao[f] = e[f]*is;
  u16* op = pre + ((size_t)b*NN + 1 + s)*CC + h*DD;
  for (int dd = 0; dd < DD; ++dd) {
    float o = 0.f;
#pragma unroll
    for (int f = 0; f < FF; ++f)
      o += e[f]*is*bf2f(xt[(((size_t)(b*SS + s))*FF + f)*CC + h*DD + dd]);
    op[dd] = f2bf(o);
  }
}

// ---------------- final projection + bias, f32 out ----------------
__global__ __launch_bounds__(128) void k_gemm_proj(const u16* __restrict__ pre,
    const u16* __restrict__ wpT, const float* __restrict__ bias,
    float* __restrict__ out) {
  const int wave = threadIdx.x >> 5;
  const int lane = threadIdx.x & 31;
  const int laneM = lane & 15, g = lane >> 4;
  long m0 = (long)blockIdx.x*64 + wave*16;
  long n0 = (long)blockIdx.y*64;
  v8f acc[4] = {vzero(),vzero(),vzero(),vzero()};
  wmma_gemm4(pre, wpT, CC, m0, (long)BB*NN - 1, n0, acc);
#pragma unroll
  for (int t = 0; t < 4; ++t) {
    int col = (int)n0 + t*16 + laneM;
    float bb = bias[col];
#pragma unroll
    for (int r = 0; r < 8; ++r) {
      long m = m0 + r + 8*g;
      if (m < (long)BB*NN) out[(size_t)m*CC + col] = acc[t][r] + bb;
    }
  }
}

extern "C" void kernel_launch(void* const* d_in, const int* in_sizes, int n_in,
                              void* d_out, int out_size, void* d_ws, size_t ws_size,
                              hipStream_t stream) {
  const float* x      = (const float*)d_in[0];
  const float* w_qkv  = (const float*)d_in[1];
  const float* w_q    = (const float*)d_in[2];
  const float* w_kv   = (const float*)d_in[3];
  const float* w_proj = (const float*)d_in[4];
  const float* b_proj = (const float*)d_in[5];
  float* out = (float*)d_out;

  char* w = (char*)d_ws;
  size_t off = 0;
  auto alloc = [&](size_t bytes) -> char* {
    char* p = w + off; off += (bytes + 255) & ~(size_t)255; return p;
  };
  u16* wqkvT = (u16*)alloc((size_t)3*CC*CC*2);
  u16* wqT   = (u16*)alloc((size_t)CC*CC*2);
  u16* wkT   = (u16*)alloc((size_t)CC*CC*2);
  u16* wpT   = (u16*)alloc((size_t)CC*CC*2);
  u16* xb    = (u16*)alloc((size_t)BB*NN*CC*2);
  u16* qh    = (u16*)alloc((size_t)BH*NN*DD*2);
  u16* khv   = (u16*)alloc((size_t)BH*NN*DD*2);
  u16* vh    = (u16*)alloc((size_t)BH*NN*DD*2);
  u16* xt    = (u16*)alloc((size_t)BB*SS*FF*CC*2);
  u16* dg    = (u16*)alloc((size_t)BB*SS*CC*2);
  u16* q2    = (u16*)alloc((size_t)BB*SS*CC*2);
  u16* k2    = (u16*)alloc((size_t)BB*SS*FF*CC*2);
  u16* pre   = (u16*)alloc((size_t)BB*NN*CC*2);

  // bf16 conversions (weights transposed to B^T row-major-K)
  k_f2bf<<<dim3(((long)BB*NN*CC + 255)/256), 256, 0, stream>>>(x, xb, (long)BB*NN*CC);
  k_transpose_bf<<<dim3((3*CC*CC + 255)/256), 256, 0, stream>>>(w_qkv, wqkvT, CC, 3*CC, 3*CC);
  k_transpose_bf<<<dim3((CC*CC + 255)/256), 256, 0, stream>>>(w_q,    wqT,   CC, CC, CC);
  k_transpose_bf<<<dim3((CC*CC + 255)/256), 256, 0, stream>>>(w_kv,   wkT,   CC, CC, 2*CC);
  k_transpose_bf<<<dim3((CC*CC + 255)/256), 256, 0, stream>>>(w_proj, wpT,   CC, CC, CC);

  k_gemm_qkv<<<dim3((BB*NN + 63)/64, (3*CC)/64), 128, 0, stream>>>(xb, wqkvT, qh, khv, vh);
  k_cls<<<dim3(BH), 256, 0, stream>>>(qh, khv, vh, pre);
  k_space_attn<<<dim3((SS + 63)/64, FF, BH), 128, 0, stream>>>(qh, khv, vh, xt);
  k_diag<<<dim3(4096), 256, 0, stream>>>(xt, dg);
  k_gemm_q2<<<dim3(BB*SS/64, CC/64), 128, 0, stream>>>(dg, wqT, q2);
  k_gemm_k2<<<dim3(BB*SS*FF/64, CC/64), 128, 0, stream>>>(xt, wkT, k2);
  k_attn2<<<dim3((BB*HH*SS + 255)/256), 256, 0, stream>>>(q2, k2, xt, pre,
                                                          out + (size_t)BB*NN*CC);
  k_gemm_proj<<<dim3((BB*NN + 63)/64, CC/64), 128, 0, stream>>>(pre, wpT, b_proj, out);

  (void)in_sizes; (void)n_in; (void)out_size; (void)ws_size;
}
